// MatchNet_48833778155888
// MI455X (gfx1250) — matrix operand order repaired and checked by reading, not compiled, not run on hardware
//
#include <hip/hip_runtime.h>

// ---------------------------------------------------------------------------
// MatchNet for MI455X (gfx1250): wave32 + WMMA f16->f32 GEMM pipeline
// ---------------------------------------------------------------------------

typedef _Float16 v8h  __attribute__((ext_vector_type(8)));
typedef _Float16 v16h __attribute__((ext_vector_type(16)));
typedef float    v8f  __attribute__((ext_vector_type(8)));

static constexpr int BB   = 4;
static constexpr int NPT  = 2048;
static constexpr int KNN  = 20;
static constexpr int DEMB = 512;
static constexpr int NH   = 4;
static constexpr int DHD  = 128;
static constexpr float NEG_INF = -1e30f;

// ---------------------------------------------------------------------------
// Generic batched GEMM:  C[r,o] = alpha * sum_k A[r,k]*Bm[o,k]  (+ bias[o])
//
// block = 128 threads = 4 waves.  Each wave owns one 16-row tile (the four
// waves cover 64 consecutive rows) and all waves share the same 64-column
// stripe -> B fragments hit L0/L2 together.  Each wave computes a 16x64 tile:
// one A fragment vs 4 B fragments -> 4 x v_wmma_f32_16x16x32_f16 per K-step.
//
// Software pipeline with *named* ping/pong register sets (no arrays, no
// dynamic indexing -> no scratch spills).  Requires K % 64 == 0 (all K in
// this pipeline are 128/512/1024/2048).  Each iteration handles two 32-wide
// K-steps; loads of one set are issued before the WMMAs of the other, so the
// pre-WMMA wait is a partial s_wait_loadcnt.
//
// Fragment layouts (16-bit):
//   A 16x32: lane<16 holds K {0..7,16..23}, lane>=16 holds K {8..15,24..31}
//            of row (lane&15).
//   B 32x16: lane<16 holds K {0..7,16..23}, lane>=16 holds K {8..15,24..31}
//            of column (lane&15).
// All loads are contiguous 16B chunks.
// ---------------------------------------------------------------------------
__global__ void __launch_bounds__(128)
gemm_wmma_kernel(const _Float16* __restrict__ A, size_t strideA, int lda,
                 const _Float16* __restrict__ Bm, size_t strideB, int ldb,
                 float* __restrict__ C, size_t strideC, int ldc,
                 const float* __restrict__ bias, float alpha, int K)
{
  const int lane = threadIdx.x & 31;
  const int wav  = threadIdx.x >> 5;        // 0..3: row tile within block
  const int tm = (blockIdx.x << 2) + wav;   // 16-row tile index
  const int to = blockIdx.y << 2;           // first 16-col tile of 64-col stripe
  const int bz = blockIdx.z;

  const _Float16* Ab = A  + (size_t)bz * strideA;
  const _Float16* Bb = Bm + (size_t)bz * strideB;
  float*          Cb = C  + (size_t)bz * strideC;

  const int nlo = lane & 15;
  const int hi  = lane >> 4;                // 0 or 1
  const int m   = (tm << 4) + nlo;          // A row
  const int kA0 = hi ? 8 : 0;
  const int kA1 = kA0 + 16;
  const int kB0 = hi ? 8 : 0;

  const _Float16* ap  = Ab + (size_t)m * lda;
  const _Float16* bp0 = Bb + (size_t)(((to + 0) << 4) + nlo) * ldb;
  const _Float16* bp1 = Bb + (size_t)(((to + 1) << 4) + nlo) * ldb;
  const _Float16* bp2 = Bb + (size_t)(((to + 2) << 4) + nlo) * ldb;
  const _Float16* bp3 = Bb + (size_t)(((to + 3) << 4) + nlo) * ldb;

  union frag { v16h v; struct { v8h lo, hi; } p; };

#define LOADA(f, kk) { (f).p.lo = *(const v8h*)(ap + (kk) + kA0); \
                       (f).p.hi = *(const v8h*)(ap + (kk) + kA1); }
#define LOADB(f, bp, kk) { (f).p.lo = *(const v8h*)((bp) + (kk) + kB0); \
                           (f).p.hi = *(const v8h*)((bp) + (kk) + kB0 + 16); }
#define WMMA(acc, fa, fb) \
  acc = __builtin_amdgcn_wmma_f32_16x16x32_f16(false, (fa).v, false, (fb).v, \
                                               (short)0, acc, false, false)

  v8f acc0 = {0.f,0.f,0.f,0.f,0.f,0.f,0.f,0.f};
  v8f acc1 = acc0, acc2 = acc0, acc3 = acc0;

  frag a0, b00, b01, b02, b03;   // even K-step set
  frag a1, b10, b11, b12, b13;   // odd  K-step set

  LOADA(a0, 0);
  LOADB(b00, bp0, 0); LOADB(b01, bp1, 0); LOADB(b02, bp2, 0); LOADB(b03, bp3, 0);

  for (int k0 = 0; k0 < K; k0 += 64) {
    // issue odd-step loads, then consume even-step fragments
    LOADA(a1, k0 + 32);
    LOADB(b10, bp0, k0 + 32); LOADB(b11, bp1, k0 + 32);
    LOADB(b12, bp2, k0 + 32); LOADB(b13, bp3, k0 + 32);
    WMMA(acc0, a0, b00); WMMA(acc1, a0, b01);
    WMMA(acc2, a0, b02); WMMA(acc3, a0, b03);
    // issue next even-step loads, then consume odd-step fragments
    if (k0 + 64 < K) {
      LOADA(a0, k0 + 64);
      LOADB(b00, bp0, k0 + 64); LOADB(b01, bp1, k0 + 64);
      LOADB(b02, bp2, k0 + 64); LOADB(b03, bp3, k0 + 64);
    }
    WMMA(acc0, a1, b10); WMMA(acc1, a1, b11);
    WMMA(acc2, a1, b12); WMMA(acc3, a1, b13);
  }
#undef LOADA
#undef LOADB
#undef WMMA

  // C/D layout: lane<16 -> rows tile+0..7 (VGPR r = row r), lane>=16 -> +8
  const int row0 = (tm << 4) + (hi ? 8 : 0);
  v8f accs[4] = {acc0, acc1, acc2, acc3};
#pragma unroll
  for (int j = 0; j < 4; ++j) {
    const int o = ((to + j) << 4) + nlo;
    const float bv = bias ? bias[o] : 0.f;
#pragma unroll
    for (int r = 0; r < 8; ++r)
      Cb[(size_t)(row0 + r) * ldc + o] = alpha * accs[j][r] + bv;
  }
}

// ---------------------------------------------------------------------------
// kNN: one thread per (b,n); top-20 of neg squared distance (insertion sort).
// ---------------------------------------------------------------------------
__global__ void knn_kernel(const float* __restrict__ x, int* __restrict__ idx)
{
  const int gid = blockIdx.x * blockDim.x + threadIdx.x;
  if (gid >= BB * NPT) return;
  const int b = gid / NPT, n = gid % NPT;
  const float* xb = x + (size_t)b * 3 * NPT;
  const float x0 = xb[n], x1 = xb[NPT + n], x2 = xb[2 * NPT + n];
  const float xxn = x0 * x0 + x1 * x1 + x2 * x2;

  float bv[KNN]; int bi[KNN];
#pragma unroll
  for (int i = 0; i < KNN; ++i) { bv[i] = NEG_INF; bi[i] = 0; }

  for (int m = 0; m < NPT; ++m) {
    const float y0 = xb[m], y1 = xb[NPT + m], y2 = xb[2 * NPT + m];
    const float inner = x0 * y0 + x1 * y1 + x2 * y2;
    const float xxm = y0 * y0 + y1 * y1 + y2 * y2;
    const float v = 2.f * inner - xxn - xxm;
    if (v > bv[KNN - 1]) {
      int p = KNN - 1;
      while (p > 0 && bv[p - 1] < v) { bv[p] = bv[p - 1]; bi[p] = bi[p - 1]; --p; }
      bv[p] = v; bi[p] = m;
    }
  }
  int* o = idx + (size_t)gid * KNN;
#pragma unroll
  for (int i = 0; i < KNN; ++i) o[i] = bi[i];
}

// ---------------------------------------------------------------------------
// Graph encoder: per point -> conv1(64) -> attnpool -> conv2(128) -> attnpool
//                -> wo(512).  One 128-thread block per (b,n), LDS staged.
// Output layout: d[(b*N+n)*512 + o]   (point-major, WMMA-friendly)
// ---------------------------------------------------------------------------
__global__ void __launch_bounds__(128)
graph_kernel(const float* __restrict__ x, const int* __restrict__ idx,
             const float* __restrict__ w1, const float* __restrict__ s1, const float* __restrict__ b1,
             const float* __restrict__ w2, const float* __restrict__ s2, const float* __restrict__ b2,
             const float* __restrict__ wo, const float* __restrict__ so, const float* __restrict__ bo,
             float* __restrict__ dout)
{
  const int n = blockIdx.x, b = blockIdx.y, t = threadIdx.x;
  const float* xb = x + (size_t)b * 3 * NPT;

  __shared__ float g[6 * KNN];
  __shared__ float h1[64 * KNN];
  __shared__ float h2[128 * KNN];
  __shared__ float xc[192];
  __shared__ int   idxs[KNN];
  __shared__ float ctr[3];

  if (t < KNN) idxs[t] = idx[((size_t)b * NPT + n) * KNN + t];
  if (t < 3)   ctr[t] = xb[t * NPT + n];
  __syncthreads();

  if (t < KNN) {
    const int j = idxs[t];
#pragma unroll
    for (int d = 0; d < 3; ++d) {
      const float f = xb[d * NPT + j];
      g[d * KNN + t]       = f - ctr[d];
      g[(3 + d) * KNN + t] = ctr[d];
    }
  }
  __syncthreads();

  for (int e = t; e < 64 * KNN; e += 128) {
    const int c = e / KNN, k = e % KNN;
    float acc = 0.f;
#pragma unroll
    for (int j = 0; j < 6; ++j) acc += w1[c * 6 + j] * g[j * KNN + k];
    acc = acc * s1[c] + b1[c];
    h1[c * KNN + k] = fmaxf(acc, 0.f);
  }
  __syncthreads();

  // h2 (reads h1) and pool1 (reads h1, writes xc[0..63]) after one barrier
  for (int e = t; e < 128 * KNN; e += 128) {
    const int c = e / KNN, k = e % KNN;
    float acc = 0.f;
    for (int j = 0; j < 64; ++j) acc += w2[c * 64 + j] * h1[j * KNN + k];
    acc = acc * s2[c] + b2[c];
    h2[c * KNN + k] = fmaxf(acc, 0.f);
  }
  if (t < 64) {
    float mx = NEG_INF;
#pragma unroll
    for (int k = 0; k < KNN; ++k) mx = fmaxf(mx, h1[t * KNN + k]);
    float se = 0.f, ws = 0.f;
#pragma unroll
    for (int k = 0; k < KNN; ++k) {
      const float h = h1[t * KNN + k];
      const float e = expf(h - mx);
      se += e; ws += h * e;
    }
    xc[t] = ws / se;
  }
  __syncthreads();

  if (t < 128) {
    float mx = NEG_INF;
#pragma unroll
    for (int k = 0; k < KNN; ++k) mx = fmaxf(mx, h2[t * KNN + k]);
    float se = 0.f, ws = 0.f;
#pragma unroll
    for (int k = 0; k < KNN; ++k) {
      const float h = h2[t * KNN + k];
      const float e = expf(h - mx);
      se += e; ws += h * e;
    }
    xc[64 + t] = ws / se;
  }
  __syncthreads();

  float* dp = dout + ((size_t)b * NPT + n) * DEMB;
  for (int o = t; o < DEMB; o += 128) {
    float acc = 0.f;
    for (int c = 0; c < 192; ++c) acc += wo[o * 192 + c] * xc[c];
    acc = acc * so[o] + bo[o];
    dp[o] = fmaxf(acc, 0.f);
  }
}

// ---------------------------------------------------------------------------
// Elementwise helpers
// ---------------------------------------------------------------------------
__global__ void f32_to_f16_kernel(const float* __restrict__ s, _Float16* __restrict__ d, size_t n)
{
  size_t i = (size_t)blockIdx.x * 256 + threadIdx.x;
  if (i < n) d[i] = (_Float16)s[i];
}
__global__ void vadd_kernel(float* __restrict__ d, const float* __restrict__ a, size_t n)
{
  size_t i = (size_t)blockIdx.x * 256 + threadIdx.x;
  if (i < n) d[i] += a[i];
}
__global__ void fill_kernel(float* __restrict__ p, float v, size_t n)
{
  size_t i = (size_t)blockIdx.x * 256 + threadIdx.x;
  if (i < n) p[i] = v;
}

// head gather/scatter: reference reshape uses channel c = d*H + h
__global__ void gather_qk_kernel(const float* __restrict__ q, _Float16* __restrict__ qh)
{
  size_t i = (size_t)blockIdx.x * 256 + threadIdx.x;          // over B*H*N*DHD
  if (i >= (size_t)BB * NH * NPT * DHD) return;
  const int d = i & (DHD - 1);
  size_t t = i >> 7;
  const int n = t & (NPT - 1);
  t >>= 11;
  const int h = t & (NH - 1);
  const int b = (int)(t >> 2);
  qh[i] = (_Float16)q[((size_t)b * NPT + n) * DEMB + d * NH + h];
}
__global__ void gather_v_kernel(const float* __restrict__ v, _Float16* __restrict__ vhT)
{
  size_t i = (size_t)blockIdx.x * 256 + threadIdx.x;          // vhT[b,h,d,m]
  if (i >= (size_t)BB * NH * DHD * NPT) return;
  const int m = i & (NPT - 1);
  size_t t = i >> 11;
  const int d = t & (DHD - 1);
  t >>= 7;
  const int h = t & (NH - 1);
  const int b = (int)(t >> 2);
  vhT[i] = (_Float16)v[((size_t)b * NPT + m) * DEMB + d * NH + h];
}
__global__ void ungather_msg_kernel(const float* __restrict__ msgh, float* __restrict__ msg)
{
  size_t i = (size_t)blockIdx.x * 256 + threadIdx.x;          // over B*N*512
  if (i >= (size_t)BB * NPT * DEMB) return;
  const int c = i & (DEMB - 1);
  size_t t = i >> 9;
  const int n = t & (NPT - 1);
  const int b = (int)(t >> 11);
  const int d = c >> 2, h = c & 3;
  msg[i] = msgh[(((size_t)b * NH + h) * NPT + n) * DHD + d];
}
__global__ void xcat_kernel(const float* __restrict__ x, const float* __restrict__ m,
                            _Float16* __restrict__ xcat)
{
  size_t i = (size_t)blockIdx.x * 256 + threadIdx.x;          // over B*N*1024
  if (i >= (size_t)BB * NPT * 2 * DEMB) return;
  const int c = i & (2 * DEMB - 1);
  const size_t r = i >> 10;
  xcat[i] = (_Float16)(c < DEMB ? x[r * DEMB + c] : m[r * DEMB + (c - DEMB)]);
}

// ---------------------------------------------------------------------------
// Softmax over attention rows -> f16 probabilities
// ---------------------------------------------------------------------------
__global__ void __launch_bounds__(256)
softmax_rows_kernel(const float* __restrict__ S, _Float16* __restrict__ P)
{
  const size_t row = blockIdx.x;
  const float* s = S + row * NPT;
  _Float16* p = P + row * NPT;
  const int tid = threadIdx.x;
  __shared__ float red[256];

  float mx = NEG_INF;
  for (int c = tid; c < NPT; c += 256) mx = fmaxf(mx, s[c]);
  red[tid] = mx; __syncthreads();
  for (int st = 128; st > 0; st >>= 1) { if (tid < st) red[tid] = fmaxf(red[tid], red[tid + st]); __syncthreads(); }
  mx = red[0]; __syncthreads();

  float sum = 0.f;
  for (int c = tid; c < NPT; c += 256) sum += expf(s[c] - mx);
  red[tid] = sum; __syncthreads();
  for (int st = 128; st > 0; st >>= 1) { if (tid < st) red[tid] += red[tid + st]; __syncthreads(); }
  const float inv = 1.f / red[0];

  for (int c = tid; c < NPT; c += 256) p[c] = (_Float16)(expf(s[c] - mx) * inv);
}

// ---------------------------------------------------------------------------
// Sinkhorn on s (B,2049,2049)
// ---------------------------------------------------------------------------
__global__ void __launch_bounds__(256)
sink_row_kernel(float* __restrict__ s)
{
  const int b = blockIdx.y, r = blockIdx.x, tid = threadIdx.x;   // r < 2048
  float* row = s + ((size_t)b * 2049 + r) * 2049;
  __shared__ float red[256];
  float mx = NEG_INF;
  for (int c = tid; c < 2049; c += 256) mx = fmaxf(mx, row[c]);
  red[tid] = mx; __syncthreads();
  for (int st = 128; st > 0; st >>= 1) { if (tid < st) red[tid] = fmaxf(red[tid], red[tid + st]); __syncthreads(); }
  mx = red[0]; __syncthreads();
  float sum = 0.f;
  for (int c = tid; c < 2049; c += 256) sum += expf(row[c] - mx);
  red[tid] = sum; __syncthreads();
  for (int st = 128; st > 0; st >>= 1) { if (tid < st) red[tid] += red[tid + st]; __syncthreads(); }
  const float lse = mx + logf(red[0]);
  for (int c = tid; c < 2049; c += 256) row[c] -= lse;
}
__global__ void __launch_bounds__(256)
sink_col_kernel(float* __restrict__ s)
{
  const int b = blockIdx.y;
  const int c = blockIdx.x * 256 + threadIdx.x;                  // c < 2048
  float* base = s + (size_t)b * 2049 * 2049;
  float mx = NEG_INF;
  for (int r = 0; r < 2049; ++r) mx = fmaxf(mx, base[(size_t)r * 2049 + c]);
  float sum = 0.f;
  for (int r = 0; r < 2049; ++r) sum += expf(base[(size_t)r * 2049 + c] - mx);
  const float lse = mx + logf(sum);
  for (int r = 0; r < 2049; ++r) base[(size_t)r * 2049 + c] -= lse;
}

// perm = exp(s[:2048,:2048]); perm_norm = perm/rowsum; weights = rowmax(perm)
__global__ void __launch_bounds__(256)
perm_kernel(const float* __restrict__ s, float* __restrict__ pn, float* __restrict__ wmax)
{
  const int b = blockIdx.y, n = blockIdx.x, tid = threadIdx.x;
  const float* row = s + ((size_t)b * 2049 + n) * 2049;
  float* orow = pn + ((size_t)b * NPT + n) * NPT;
  __shared__ float red[256], red2[256];
  float sum = 0.f, mx = NEG_INF;
  for (int m = tid; m < NPT; m += 256) {
    const float e = expf(row[m]);
    orow[m] = e; sum += e; mx = fmaxf(mx, e);
  }
  red[tid] = sum; red2[tid] = mx; __syncthreads();
  for (int st = 128; st > 0; st >>= 1) {
    if (tid < st) { red[tid] += red[tid + st]; red2[tid] = fmaxf(red2[tid], red2[tid + st]); }
    __syncthreads();
  }
  const float inv = 1.f / (red[0] + 1e-8f);
  for (int m = tid; m < NPT; m += 256) orow[m] *= inv;
  if (tid == 0) wmax[b * NPT + n] = red2[0];
}

// weighted_tgt[b,d,n] = sum_m tgt[b,d,m] * perm_norm[b,n,m]
__global__ void __launch_bounds__(256)
wtgt_kernel(const float* __restrict__ tgt, const float* __restrict__ pn, float* __restrict__ wt)
{
  const int b = blockIdx.y, n = blockIdx.x, tid = threadIdx.x;
  const float* prow = pn + ((size_t)b * NPT + n) * NPT;
  const float* tb = tgt + (size_t)b * 3 * NPT;
  __shared__ float red[256];
  float a0 = 0.f, a1 = 0.f, a2 = 0.f;
  for (int m = tid; m < NPT; m += 256) {
    const float p = prow[m];
    a0 += p * tb[m]; a1 += p * tb[NPT + m]; a2 += p * tb[2 * NPT + m];
  }
  float acc[3] = {a0, a1, a2};
#pragma unroll
  for (int d = 0; d < 3; ++d) {
    red[tid] = acc[d]; __syncthreads();
    for (int st = 128; st > 0; st >>= 1) { if (tid < st) red[tid] += red[tid + st]; __syncthreads(); }
    if (tid == 0) wt[(size_t)b * 3 * NPT + d * NPT + n] = red[0];
    __syncthreads();
  }
}

__device__ __forceinline__ float block_reduce_sum256(float* red, float v, int tid)
{
  red[tid] = v; __syncthreads();
  for (int st = 128; st > 0; st >>= 1) { if (tid < st) red[tid] += red[tid + st]; __syncthreads(); }
  const float r = red[0]; __syncthreads();
  return r;
}

// per-batch centroids + 3x3 weighted covariance -> stats[b*16 + {sc,tc,H}]
__global__ void __launch_bounds__(256)
centroid_kernel(const float* __restrict__ src, const float* __restrict__ wt,
                const float* __restrict__ weights, float* __restrict__ stats)
{
  const int b = blockIdx.x, tid = threadIdx.x;
  __shared__ float red[256];
  const float* w = weights + b * NPT;
  const float* sb = src + (size_t)b * 3 * NPT;
  const float* tb = wt + (size_t)b * 3 * NPT;

  float acc = 0.f;
  for (int n = tid; n < NPT; n += 256) acc += w[n];
  const float wsum = block_reduce_sum256(red, acc, tid);
  const float inv = 1.f / (wsum + 1e-8f);

  float sc[3], tc[3];
  for (int d = 0; d < 3; ++d) {
    acc = 0.f; for (int n = tid; n < NPT; n += 256) acc += sb[d * NPT + n] * w[n];
    sc[d] = block_reduce_sum256(red, acc, tid) * inv;
    acc = 0.f; for (int n = tid; n < NPT; n += 256) acc += tb[d * NPT + n] * w[n];
    tc[d] = block_reduce_sum256(red, acc, tid) * inv;
  }
  for (int d = 0; d < 3; ++d)
    for (int e = 0; e < 3; ++e) {
      acc = 0.f;
      for (int n = tid; n < NPT; n += 256)
        acc += (sb[d * NPT + n] - sc[d]) * (tb[e * NPT + n] - tc[e]) * (w[n] * inv);
      const float h = block_reduce_sum256(red, acc, tid);
      if (tid == 0) stats[b * 16 + 6 + d * 3 + e] = h;
    }
  if (tid == 0) {
#pragma unroll
    for (int d = 0; d < 3; ++d) { stats[b * 16 + d] = sc[d]; stats[b * 16 + 3 + d] = tc[d]; }
  }
}

// 3x3 SVD (Jacobi on H^T H) -> R, t per batch.  One thread per batch.
__global__ void svd_kernel(const float* __restrict__ stats, float* __restrict__ out)
{
  const int b = threadIdx.x;
  if (b >= BB) return;
  const float* st = stats + b * 16;
  float sc[3] = {st[0], st[1], st[2]}, tc[3] = {st[3], st[4], st[5]};
  float Hm[9];
#pragma unroll
  for (int i = 0; i < 9; ++i) Hm[i] = st[6 + i];

  float A[9];
  for (int i = 0; i < 3; ++i)
    for (int j = 0; j < 3; ++j) {
      float s = 0.f;
      for (int k = 0; k < 3; ++k) s += Hm[k * 3 + i] * Hm[k * 3 + j];
      A[i * 3 + j] = s;
    }
  float V[9] = {1, 0, 0, 0, 1, 0, 0, 0, 1};
  const int PP[3] = {0, 0, 1}, QQ[3] = {1, 2, 2};
  for (int sweep = 0; sweep < 30; ++sweep)
    for (int pr = 0; pr < 3; ++pr) {
      const int p = PP[pr], q = QQ[pr];
      const float apq = A[p * 3 + q];
      if (fabsf(apq) < 1e-20f) continue;
      const float theta = (A[q * 3 + q] - A[p * 3 + p]) / (2.f * apq);
      const float t = copysignf(1.f, theta) / (fabsf(theta) + sqrtf(theta * theta + 1.f));
      const float c = 1.f / sqrtf(t * t + 1.f), s = t * c;
      for (int k = 0; k < 3; ++k) {
        const float akp = A[k * 3 + p], akq = A[k * 3 + q];
        A[k * 3 + p] = c * akp - s * akq; A[k * 3 + q] = s * akp + c * akq;
      }
      for (int k = 0; k < 3; ++k) {
        const float apk = A[p * 3 + k], aqk = A[q * 3 + k];
        A[p * 3 + k] = c * apk - s * aqk; A[q * 3 + k] = s * apk + c * aqk;
      }
      for (int k = 0; k < 3; ++k) {
        const float vkp = V[k * 3 + p], vkq = V[k * 3 + q];
        V[k * 3 + p] = c * vkp - s * vkq; V[k * 3 + q] = s * vkp + c * vkq;
      }
    }
  float ev[3] = {A[0], A[4], A[8]};
  int ord[3] = {0, 1, 2};
  for (int i = 0; i < 2; ++i)
    for (int j = 0; j < 2 - i; ++j)
      if (ev[ord[j]] < ev[ord[j + 1]]) { int tmp = ord[j]; ord[j] = ord[j + 1]; ord[j + 1] = tmp; }

  float Vv[9], sv[3];
  for (int j = 0; j < 3; ++j) {
    for (int i = 0; i < 3; ++i) Vv[i * 3 + j] = V[i * 3 + ord[j]];
    sv[j] = sqrtf(fmaxf(ev[ord[j]], 0.f));
  }
  float U[9];
  for (int j = 0; j < 3; ++j) {
    float hv[3];
    for (int i = 0; i < 3; ++i)
      hv[i] = Hm[i * 3 + 0] * Vv[0 * 3 + j] + Hm[i * 3 + 1] * Vv[1 * 3 + j] + Hm[i * 3 + 2] * Vv[2 * 3 + j];
    if (j < 2 || sv[j] > 1e-9f * (sv[0] + 1e-20f)) {
      const float invs = 1.f / (sv[j] + 1e-20f);
      for (int i = 0; i < 3; ++i) U[i * 3 + j] = hv[i] * invs;
    } else {
      U[0 * 3 + 2] = U[1 * 3 + 0] * U[2 * 3 + 1] - U[2 * 3 + 0] * U[1 * 3 + 1];
      U[1 * 3 + 2] = U[2 * 3 + 0] * U[0 * 3 + 1] - U[0 * 3 + 0] * U[2 * 3 + 1];
      U[2 * 3 + 2] = U[0 * 3 + 0] * U[1 * 3 + 1] - U[1 * 3 + 0] * U[0 * 3 + 1];
    }
  }
  float M[9];
  for (int i = 0; i < 3; ++i)
    for (int j = 0; j < 3; ++j) {
      float s = 0.f;
      for (int k = 0; k < 3; ++k) s += Vv[i * 3 + k] * U[j * 3 + k];
      M[i * 3 + j] = s;
    }
  const float det = M[0] * (M[4] * M[8] - M[5] * M[7]) - M[1] * (M[3] * M[8] - M[5] * M[6])
                  + M[2] * (M[3] * M[7] - M[4] * M[6]);
  float R[9];
  for (int i = 0; i < 3; ++i)
    for (int j = 0; j < 3; ++j) {
      float s = 0.f;
      for (int k = 0; k < 3; ++k) s += Vv[i * 3 + k] * ((k == 2) ? det : 1.f) * U[j * 3 + k];
      R[i * 3 + j] = s;
    }
  for (int i = 0; i < 9; ++i) out[b * 9 + i] = R[i];
  for (int i = 0; i < 3; ++i)
    out[36 + b * 3 + i] = tc[i] - (R[i * 3 + 0] * sc[0] + R[i * 3 + 1] * sc[1] + R[i * 3 + 2] * sc[2]);
}

// ---------------------------------------------------------------------------
// Host orchestration
// ---------------------------------------------------------------------------
static inline size_t cdiv(size_t a, size_t b) { return (a + b - 1) / b; }

extern "C" void kernel_launch(void* const* d_in, const int* in_sizes, int n_in,
                              void* d_out, int out_size, void* d_ws, size_t ws_size,
                              hipStream_t stream)
{
  (void)in_sizes; (void)n_in; (void)out_size; (void)ws_size;
  const float* src  = (const float*)d_in[0];
  const float* tgt  = (const float*)d_in[1];
  const float* g_w1 = (const float*)d_in[2];
  const float* g_s1 = (const float*)d_in[3];
  const float* g_b1 = (const float*)d_in[4];
  const float* g_w2 = (const float*)d_in[5];
  const float* g_s2 = (const float*)d_in[6];
  const float* g_b2 = (const float*)d_in[7];
  const float* g_wo = (const float*)d_in[8];
  const float* g_so = (const float*)d_in[9];
  const float* g_bo = (const float*)d_in[10];
  const float* q_w  = (const float*)d_in[11];
  const float* q_b  = (const float*)d_in[12];
  const float* k_w  = (const float*)d_in[13];
  const float* k_b  = (const float*)d_in[14];
  const float* v_w  = (const float*)d_in[15];
  const float* v_b  = (const float*)d_in[16];
  const float* m_w  = (const float*)d_in[17];
  const float* m_b  = (const float*)d_in[18];
  const float* mlp_w = (const float*)d_in[19];
  float* out = (float*)d_out;

  // ---- workspace carving (bump allocator, 256B aligned) ----
  char* wsp = (char*)d_ws;
  auto carve = [&](size_t bytes) -> void* {
    void* p = (void*)wsp;
    wsp += (bytes + 255) & ~(size_t)255;
    return p;
  };
  const size_t ROWS = (size_t)BB * NPT;                 // 8192
  int*       idx0  = (int*)carve(ROWS * KNN * 4);
  int*       idx1  = (int*)carve(ROWS * KNN * 4);
  float*     dA    = (float*)carve(ROWS * DEMB * 4);
  float*     dBv   = (float*)carve(ROWS * DEMB * 4);
  _Float16*  dAh   = (_Float16*)carve(ROWS * DEMB * 2);
  _Float16*  dBh   = (_Float16*)carve(ROWS * DEMB * 2);
  _Float16*  qw16  = (_Float16*)carve((size_t)DEMB * DEMB * 2);
  _Float16*  kw16  = (_Float16*)carve((size_t)DEMB * DEMB * 2);
  _Float16*  vw16  = (_Float16*)carve((size_t)DEMB * DEMB * 2);
  _Float16*  mw16  = (_Float16*)carve((size_t)DEMB * DEMB * 2);
  _Float16*  mlpw16 = (_Float16*)carve((size_t)DEMB * 2 * DEMB * 2);
  float*     q_out = (float*)carve(ROWS * DEMB * 4);
  float*     k_out = (float*)carve(ROWS * DEMB * 4);
  float*     v_out = (float*)carve(ROWS * DEMB * 4);
  _Float16*  qh    = (_Float16*)carve((size_t)BB * NH * NPT * DHD * 2);
  _Float16*  kh    = (_Float16*)carve((size_t)BB * NH * NPT * DHD * 2);
  _Float16*  vhT   = (_Float16*)carve((size_t)BB * NH * DHD * NPT * 2);
  float*     S     = (float*)carve((size_t)BB * NH * NPT * NPT * 4);   // 268MB
  _Float16*  P     = (_Float16*)carve((size_t)BB * NH * NPT * NPT * 2);
  float*     msgh  = (float*)carve((size_t)BB * NH * NPT * DHD * 4);
  float*     msgf  = (float*)carve(ROWS * DEMB * 4);
  _Float16*  msg16 = (_Float16*)carve(ROWS * DEMB * 2);
  float*     m_out = (float*)carve(ROWS * DEMB * 4);
  _Float16*  xcat  = (_Float16*)carve(ROWS * 2 * DEMB * 2);
  float*     prop0 = (float*)carve(ROWS * DEMB * 4);
  float*     prop1 = (float*)carve(ROWS * DEMB * 4);
  float*     wt    = (float*)carve((size_t)BB * 3 * NPT * 4);
  float*     weights = (float*)carve((size_t)BB * NPT * 4);
  float*     stats = (float*)carve(BB * 16 * 4);
  float*     sink  = S;                                 // alias: S unused after attention

  const float inv_sqrt_dh = 0.08838834764831845f;       // 1/sqrt(128)
  const float inv_sqrt_d  = 0.044194173824159216f;      // 1/sqrt(512)

  // block = 4 waves; wave tile 16x64 -> block tile 64x64
  auto gemm = [&](const _Float16* Ap, size_t sA, int lda,
                  const _Float16* Bp, size_t sB, int ldb,
                  float* Cp, size_t sC, int ldc,
                  const float* bias, float alpha, int M, int O, int K, int nb) {
    dim3 grid(M / 64, O / 64, nb);
    gemm_wmma_kernel<<<grid, dim3(128), 0, stream>>>(Ap, sA, lda, Bp, sB, ldb,
                                                     Cp, sC, ldc, bias, alpha, K);
  };
  auto tof16 = [&](const float* s, _Float16* d, size_t n) {
    f32_to_f16_kernel<<<(unsigned)cdiv(n, 256), 256, 0, stream>>>(s, d, n);
  };

  // ---- stage 1: kNN + graph encoder ----
  knn_kernel<<<(unsigned)cdiv(ROWS, 256), 256, 0, stream>>>(src, idx0);
  knn_kernel<<<(unsigned)cdiv(ROWS, 256), 256, 0, stream>>>(tgt, idx1);
  graph_kernel<<<dim3(NPT, BB), 128, 0, stream>>>(src, idx0, g_w1, g_s1, g_b1,
                                                  g_w2, g_s2, g_b2, g_wo, g_so, g_bo, dA);
  graph_kernel<<<dim3(NPT, BB), 128, 0, stream>>>(tgt, idx1, g_w1, g_s1, g_b1,
                                                  g_w2, g_s2, g_b2, g_wo, g_so, g_bo, dBv);

  // ---- weights + activations to f16 ----
  tof16(q_w, qw16, (size_t)DEMB * DEMB);
  tof16(k_w, kw16, (size_t)DEMB * DEMB);
  tof16(v_w, vw16, (size_t)DEMB * DEMB);
  tof16(m_w, mw16, (size_t)DEMB * DEMB);
  tof16(mlp_w, mlpw16, (size_t)DEMB * 2 * DEMB);
  tof16(dA, dAh, ROWS * DEMB);
  tof16(dBv, dBh, ROWS * DEMB);

  // ---- stage 2: cross-attention propagation (both directions) ----
  const size_t hcnt = (size_t)BB * NH * NPT * DHD;
  auto prop = [&](const _Float16* xh, const _Float16* srch, const float* xf, float* pout) {
    gemm(xh, 0, DEMB, qw16, 0, DEMB, q_out, 0, DEMB, q_b, 1.f, (int)ROWS, DEMB, DEMB, 1);
    gemm(srch, 0, DEMB, kw16, 0, DEMB, k_out, 0, DEMB, k_b, 1.f, (int)ROWS, DEMB, DEMB, 1);
    gemm(srch, 0, DEMB, vw16, 0, DEMB, v_out, 0, DEMB, v_b, 1.f, (int)ROWS, DEMB, DEMB, 1);
    gather_qk_kernel<<<(unsigned)cdiv(hcnt, 256), 256, 0, stream>>>(q_out, qh);
    gather_qk_kernel<<<(unsigned)cdiv(hcnt, 256), 256, 0, stream>>>(k_out, kh);
    gather_v_kernel<<<(unsigned)cdiv(hcnt, 256), 256, 0, stream>>>(v_out, vhT);
    // scores[b,h] = qh (N x 128) x kh^T  -> S (N x N), scaled
    gemm(qh, (size_t)NPT * DHD, DHD, kh, (size_t)NPT * DHD, DHD,
         S, (size_t)NPT * NPT, NPT, nullptr, inv_sqrt_dh, NPT, NPT, DHD, BB * NH);
    softmax_rows_kernel<<<BB * NH * NPT, 256, 0, stream>>>(S, P);
    // msgh[b,h] = P (N x N) x vhT^T -> (N x 128)
    gemm(P, (size_t)NPT * NPT, NPT, vhT, (size_t)DHD * NPT, NPT,
         msgh, (size_t)NPT * DHD, DHD, nullptr, 1.f, NPT, DHD, NPT, BB * NH);
    ungather_msg_kernel<<<(unsigned)cdiv(ROWS * DEMB, 256), 256, 0, stream>>>(msgh, msgf);
    tof16(msgf, msg16, ROWS * DEMB);
    gemm(msg16, 0, DEMB, mw16, 0, DEMB, m_out, 0, DEMB, m_b, 1.f, (int)ROWS, DEMB, DEMB, 1);
    xcat_kernel<<<(unsigned)cdiv(ROWS * 2 * DEMB, 256), 256, 0, stream>>>(xf, m_out, xcat);
    gemm(xcat, 0, 2 * DEMB, mlpw16, 0, 2 * DEMB, pout, 0, DEMB, nullptr, 1.f,
         (int)ROWS, DEMB, 2 * DEMB, 1);
  };
  prop(dAh, dBh, dA, prop0);
  prop(dBh, dAh, dBv, prop1);
  vadd_kernel<<<(unsigned)cdiv(ROWS * DEMB, 256), 256, 0, stream>>>(dA, prop0, ROWS * DEMB);
  vadd_kernel<<<(unsigned)cdiv(ROWS * DEMB, 256), 256, 0, stream>>>(dBv, prop1, ROWS * DEMB);
  tof16(dA, dAh, ROWS * DEMB);
  tof16(dBv, dBh, ROWS * DEMB);

  // ---- stage 3: dists + Sinkhorn ----
  const size_t sinkN = (size_t)BB * 2049 * 2049;
  fill_kernel<<<(unsigned)cdiv(sinkN, 256), 256, 0, stream>>>(sink, 0.f, sinkN);
  gemm(dAh, (size_t)NPT * DEMB, DEMB, dBh, (size_t)NPT * DEMB, DEMB,
       sink, (size_t)2049 * 2049, 2049, nullptr, inv_sqrt_d, NPT, NPT, DEMB, BB);
  for (int it = 0; it < 5; ++it) {
    sink_row_kernel<<<dim3(NPT, BB), 256, 0, stream>>>(sink);
    sink_col_kernel<<<dim3(NPT / 256, BB), 256, 0, stream>>>(sink);
  }

  // ---- stage 4: perm_norm, weighted Kabsch ----
  float* permn = out + 48;                               // [R(36) | t(12) | perm_norm]
  perm_kernel<<<dim3(NPT, BB), 256, 0, stream>>>(sink, permn, weights);
  wtgt_kernel<<<dim3(NPT, BB), 256, 0, stream>>>(tgt, permn, wt);
  centroid_kernel<<<BB, 256, 0, stream>>>(src, wt, weights, stats);
  svd_kernel<<<1, BB, 0, stream>>>(stats, out);
}
